// AttentionBlock_84524956385733
// MI455X (gfx1250) — matrix operand order repaired and checked by reading, not compiled, not run on hardware
//
#include <hip/hip_runtime.h>

// MI455X / gfx1250, wave32. Matrix math via v_wmma_f32_16x16x32_f16.
// R-tile staging via double-buffered async global->LDS (ASYNCcnt),
// per CDNA5 ISA.

typedef __attribute__((ext_vector_type(16))) _Float16 v16h;
typedef __attribute__((ext_vector_type(8)))  _Float16 v8h;
typedef __attribute__((ext_vector_type(8)))  float    v8f;
typedef __attribute__((ext_vector_type(4)))  float    v4f;

#define NB   4
#define NH   8
#define NT   2048
#define NIN  256
#define NOUT 256

union AF { v16h v; v8h half8[2]; _Float16 h[16]; };
union CF { v8f v; float f[8]; };

__device__ __forceinline__ int lane_id() { return (int)(threadIdx.x & 31); }

__device__ __forceinline__ v8f zero8() {
  v8f z = {0.f, 0.f, 0.f, 0.f, 0.f, 0.f, 0.f, 0.f};
  return z;
}

__device__ __forceinline__ v8f wmma_f16(v16h a, v16h b, v8f c) {
  return __builtin_amdgcn_wmma_f32_16x16x32_f16(
      /*neg_a=*/false, a, /*neg_b=*/false, b,
      /*c_mod=*/(short)0, c, /*reuse_a=*/false, /*reuse_b=*/false);
}

// ---- Fragment loaders (ISA 7.12.2 layouts, 128-bit vectorized) -------------
// A fragment 16x32 f16: lane m=lane&15 holds row m; two contiguous 8-half
// runs at K = kb and kb+16  (kb = 8*(lane>>4)).
__device__ __forceinline__ v16h load_a_f16(const _Float16* src, int ld) {
  int lane = lane_id();
  int m  = lane & 15;
  int kb = (lane >> 4) << 3;
  const _Float16* p = src + (size_t)m * ld + kb;
  AF a;
  a.half8[0] = *(const v8h*)(p);
  a.half8[1] = *(const v8h*)(p + 16);
  return a.v;
}

// B fragment 32x16 where B(k,n) = src[n*ld + k] (rows of src become columns).
// Per lane: two contiguous 8-half runs at row n, K = kb and kb+16 (kb = 8*(lane>>4)).
__device__ __forceinline__ v16h load_b_rg_f16(const _Float16* src, int ld) {
  int lane = lane_id();
  int n  = lane & 15;
  int kb = (lane >> 4) << 3;
  const _Float16* p = src + (size_t)n * ld + kb;
  AF b;
  b.half8[0] = *(const v8h*)(p);
  b.half8[1] = *(const v8h*)(p + 16);
  return b.v;
}

// ---- Async global->LDS staging: one 64x256 f16 tile (32 KB), 16B/thread x8 -
__device__ __forceinline__ void stage_tile_async(const _Float16* gsrc, _Float16* lbase) {
#pragma unroll
  for (int c = 0; c < 8; ++c) {
    int chunk = (int)threadIdx.x + c * 256;               // 16-byte chunk id
    unsigned loff = (unsigned)(size_t)&(((v8h*)lbase)[chunk]);
    const _Float16* gp = gsrc + (size_t)chunk * 8;
    asm volatile("global_load_async_to_lds_b128 %0, %1, off"
                 :: "v"(loff), "v"(gp) : "memory");
  }
}

// ---- Pass 0a: generic f32 -> f16 copy (128-bit I/O) ------------------------
__global__ void cvt_f16(const float* __restrict__ src, _Float16* __restrict__ dst, int n8) {
  int idx = (int)(blockIdx.x * blockDim.x + threadIdx.x);
  int stride = (int)(gridDim.x * blockDim.x);
  for (int i = idx; i < n8; i += stride) {
    v4f a = ((const v4f*)src)[2 * i];
    v4f b = ((const v4f*)src)[2 * i + 1];
    v8h o;
#pragma unroll
    for (int j = 0; j < 4; ++j) {
      o[j]     = (_Float16)a[j];
      o[j + 4] = (_Float16)b[j];
    }
    ((v8h*)dst)[i] = o;
  }
}

// ---- Pass 0b: Qt[h][j][i] = (f16) Q[h][i][j] (tiny, once) ------------------
__global__ void cvt_qt(const float* __restrict__ Q, _Float16* __restrict__ Qt) {
  int h = (int)blockIdx.z;
  int i = (int)(blockIdx.y * 16 + threadIdx.y);
  int j = (int)(blockIdx.x * 16 + threadIdx.x);
  Qt[((size_t)h * NIN + j) * NIN + i] = (_Float16)Q[((size_t)h * NIN + i) * NIN + j];
}

// ---- Pass 1: RQ[b,h] = R @ Q[h] (row-major f16); Vt[b,h] = (R @ E[h]^T)^T --
// grid: (NT/16, NB*NH, 2), block 256 (8 waves); wave w does j-tiles {2w,2w+1}.
// Both B operands are f16 row-gathers: mode 0 from Qt, mode 1 from Eh.
__global__ void proj_kernel(const _Float16* __restrict__ Rh, const _Float16* __restrict__ Qt,
                            const _Float16* __restrict__ Eh, _Float16* __restrict__ RQ,
                            _Float16* __restrict__ Vt) {
  int t0   = (int)blockIdx.x * 16;
  int bh   = (int)blockIdx.y;
  int h    = bh & 7;
  int b    = bh >> 3;
  int mode = (int)blockIdx.z;               // 0: RQ via Qt, 1: Vt via Eh
  int w    = (int)(threadIdx.x >> 5);

  const _Float16* Rb = Rh + (size_t)b * NT * NIN;
  const _Float16* W  = (mode == 0 ? Qt : Eh) + (size_t)h * NIN * NIN;

#pragma unroll
  for (int jt = 0; jt < 2; ++jt) {
    int j0 = (w * 2 + jt) * 16;
    CF c; c.v = zero8();
#pragma unroll
    for (int k = 0; k < NIN; k += 32) {
      v16h a  = load_a_f16(Rb + (size_t)t0 * NIN + k, NIN);
      v16h bb = load_b_rg_f16(W + (size_t)j0 * NIN + k, NIN);  // B(k,n)=W[j0+n][k]
      c.v = wmma_f16(a, bb, c.v);
    }
    int lane = lane_id();
    int n  = lane & 15;
    int mb = (lane >> 4) << 3;
    if (mode == 0) {
      _Float16* dst = RQ + (size_t)bh * NT * NIN;
#pragma unroll
      for (int rr = 0; rr < 8; ++rr)
        dst[(size_t)(t0 + mb + rr) * NIN + j0 + n] = (_Float16)c.f[rr];
    } else {
      // transposed store: Vt[i][u]; per lane 8 consecutive u -> one 16B store
      _Float16* dst = Vt + (size_t)bh * NOUT * NT;
      v8h o;
#pragma unroll
      for (int rr = 0; rr < 8; ++rr) o[rr] = (_Float16)c.f[rr];
      *(v8h*)&dst[(size_t)(j0 + n) * NT + t0 + mb] = o;
    }
  }
}

// ---- Pass 2: out[b] = sum_h tril(RQ[b,h] @ R^T) @ V[b,h] -------------------
// grid: (NT/32, NB), block 256 = 8 waves as 2(t-half) x 4(16-u / 64-i slice).
// u-outer / h-inner; R u-tiles double-buffered through async global->LDS so
// the next tile's DMA overlaps the current tile's 8-head WMMA work.
__global__ void attn_kernel(const _Float16* __restrict__ RQ, const _Float16* __restrict__ Vt,
                            const _Float16* __restrict__ Rh, float* __restrict__ out) {
  __shared__ _Float16 Rtile[2][64][NIN];    // 64 KB: double-buffered R rows
  __shared__ _Float16 Smat[2][16][64];      // 4 KB: masked f16 score tiles

  int tt = (int)blockIdx.x;                 // 32 t-rows per block
  int b  = (int)blockIdx.y;
  int w  = (int)(threadIdx.x >> 5);
  int mt = w >> 2;                          // t-half: 0/1
  int ni = w & 3;                           // u-quarter / i-slice: 0..3
  int lane = lane_id();

  int t0   = tt * 32 + mt * 16;
  int i0   = ni * 64;
  int t_hi = tt * 32 + 31;
  const _Float16* Rb = Rh + (size_t)b * NT * NIN;

  CF acc[4];
#pragma unroll
  for (int it = 0; it < 4; ++it) acc[it].v = zero8();

  int nsteps = t_hi / 64 + 1;
  stage_tile_async(Rb, &Rtile[0][0][0]);    // prologue: tile 0 in flight
  int cur = 0;

  for (int si = 0; si < nsteps; ++si) {
    int u0 = si * 64;
    if (si + 1 < nsteps) {
      // overlap: launch next tile's DMA, then wait only for the current tile
      stage_tile_async(Rb + (size_t)(u0 + 64) * NIN, &Rtile[cur ^ 1][0][0]);
      asm volatile("s_wait_asynccnt 8" ::: "memory");   // in-order: cur done
    } else {
      asm volatile("s_wait_asynccnt 0" ::: "memory");
    }
    __syncthreads();

    for (int h = 0; h < NH; ++h) {
      const _Float16* RQbh = RQ + (size_t)(b * NH + h) * NT * NIN;
      const _Float16* Vtbh = Vt + (size_t)(b * NH + h) * NOUT * NT;

      // --- score subtile S[t0..+16, u0+ni*16..+16] = RQ @ R^T  (K = 256)
      v8f s = zero8();
#pragma unroll
      for (int k = 0; k < NIN; k += 32) {
        v16h a  = load_a_f16(RQbh + (size_t)t0 * NIN + k, NIN);
        v16h bb = load_b_rg_f16(&Rtile[cur][ni * 16][k], NIN);
        s = wmma_f16(a, bb, s);
      }
      // causal mask in the f32 accumulator, stash as f16 A-matrix in LDS
      {
        CF sc; sc.v = s;
        int n  = lane & 15;
        int mb = (lane >> 4) << 3;
        int uc0 = u0 + ni * 16;
#pragma unroll
        for (int rr = 0; rr < 8; ++rr) {
          int m = mb + rr;
          float val = ((uc0 + n) <= (t0 + m)) ? sc.f[rr] : 0.f;
          Smat[mt][m][ni * 16 + n] = (_Float16)val;
        }
      }
      __syncthreads();
      // --- O(16x64) += S(16x64) @ V(64 x 64-slice); B from transposed Vt
#pragma unroll
      for (int kk = 0; kk < 64; kk += 32) {
        v16h aS = load_a_f16(&Smat[mt][0][kk], 64);
#pragma unroll
        for (int it = 0; it < 4; ++it) {
          v16h bb = load_b_rg_f16(Vtbh + (size_t)(i0 + it * 16) * NT + u0 + kk, NT);
          acc[it].v = wmma_f16(aS, bb, acc[it].v);
        }
      }
      __syncthreads();                       // Smat / Rtile reused next iter
    }
    cur ^= 1;
  }

  // epilogue: single store of the h-summed tile
  {
    int n  = lane & 15;
    int mb = (lane >> 4) << 3;
#pragma unroll
    for (int it = 0; it < 4; ++it)
#pragma unroll
      for (int rr = 0; rr < 8; ++rr)
        out[((size_t)b * NT + t0 + mb + rr) * NOUT + i0 + it * 16 + n] = acc[it].f[rr];
  }
}

extern "C" void kernel_launch(void* const* d_in, const int* in_sizes, int n_in,
                              void* d_out, int out_size, void* d_ws, size_t ws_size,
                              hipStream_t stream) {
  const float* r = (const float*)d_in[0];   // [1,4,2048,256]
  const float* Q = (const float*)d_in[1];   // [1,8,256,256]
  const float* E = (const float*)d_in[2];   // [1,8,256,256]
  float* out     = (float*)d_out;           // [1,4,2048,256]

  const size_t per  = (size_t)NB * NH * NT * NIN;  // 16.78M halfs
  const size_t wsz  = (size_t)NH * NIN * NIN;      // 512K halfs
  _Float16* RQ = (_Float16*)d_ws;                  // 33.5 MB, row-major [t][j]
  _Float16* Vt = RQ + per;                         // 33.5 MB, transposed [i][u]
  _Float16* Rh = Vt + per;                         // 4 MB, f16 copy of R
  _Float16* Qt = Rh + (size_t)NB * NT * NIN;       // 1 MB, f16 Q^T
  _Float16* Eh = Qt + wsz;                         // 1 MB, f16 E

  hipLaunchKernelGGL(cvt_f16, dim3(512), dim3(256), 0, stream,
                     r, Rh, NB * NT * NIN / 8);
  hipLaunchKernelGGL(cvt_f16, dim3(256), dim3(256), 0, stream,
                     E, Eh, NH * NIN * NIN / 8);
  hipLaunchKernelGGL(cvt_qt, dim3(NIN / 16, NIN / 16, NH), dim3(16, 16), 0, stream,
                     Q, Qt);
  hipLaunchKernelGGL(proj_kernel, dim3(NT / 16, NB * NH, 2), dim3(256), 0, stream,
                     Rh, Qt, Eh, RQ, Vt);
  hipLaunchKernelGGL(attn_kernel, dim3(NT / 32, NB), dim3(256), 0, stream,
                     RQ, Vt, Rh, out);
}
